// ResBlock_59605556133920
// MI455X (gfx1250) — matrix operand order repaired and checked by reading, not compiled, not run on hardware
//
#include <hip/hip_runtime.h>
#include <cstdint>
#include <cstddef>

typedef __attribute__((ext_vector_type(16))) __bf16 v16bf;
typedef __attribute__((ext_vector_type(8)))  float  v8f;

union Frag { v16bf v; uint4 q[2]; };

__device__ __forceinline__ unsigned short f2bf(float f) {
  unsigned int u = __float_as_uint(f);
  u += 0x7FFFu + ((u >> 16) & 1u);
  return (unsigned short)(u >> 16);
}
__device__ __forceinline__ float siluf(float x) {
  return x / (1.0f + __expf(-x));
}

// ---------------- small prep kernels ----------------

__global__ void k_copy(const float4* __restrict__ src, float4* __restrict__ dst, int n) {
  int i = blockIdx.x * 256 + threadIdx.x;
  if (i < n) dst[i] = src[i];
}

// Re-layout conv weights (Co,C,3,3) f32 -> Wt[tap][co][c] bf16, tap = ky*3+kx
__global__ void k_prep_w(const float* __restrict__ w_in, const float* __restrict__ w_out,
                         unsigned short* __restrict__ wt_in, unsigned short* __restrict__ wt_out) {
  int j = blockIdx.x * 256 + threadIdx.x;           // 0..294911
  int sel = j >= 147456;
  int jj = j - sel * 147456;
  int tap  = jj >> 14;
  int rest = jj & 16383;
  int co = rest >> 7;
  int c  = rest & 127;
  int ky = tap / 3, kx = tap - ky * 3;
  const float* src = sel ? w_out : w_in;
  float v = src[(co * 128 + c) * 9 + ky * 3 + kx];
  unsigned short* dst = sel ? wt_out : wt_in;
  dst[jj] = f2bf(v);
}

// emb_out[n][co] = silu(emb[n]) . emb_w[co] + emb_b[co]
__global__ void k_emb(const float* __restrict__ emb, const float* __restrict__ emb_w,
                      const float* __restrict__ emb_b, float* __restrict__ out) {
  int t = blockIdx.x * 256 + threadIdx.x;  // 2048
  int n = t >> 7; int co = t & 127;
  float acc = emb_b[co];
  const float* e = emb + n * 512;
  const float* w = emb_w + co * 512;
  for (int i = 0; i < 512; ++i) acc += siluf(e[i]) * w[i];
  out[t] = acc;
}

// bilinear resize dense_emb (16,64,36,36) -> (16,64,10,10)
__global__ void k_resize(const float* __restrict__ de, float* __restrict__ out) {
  int t = blockIdx.x * 256 + threadIdx.x;  // 102400
  int pix = t % 100; int rc = t / 100;     // rc = n*64+cd
  int oy = pix / 10, ox = pix - (pix / 10) * 10;
  float ys = (float)(oy * 35) / 9.0f;
  float xs = (float)(ox * 35) / 9.0f;
  int y0 = (int)ys; if (y0 > 34) y0 = 34;
  int x0 = (int)xs; if (x0 > 34) x0 = 34;
  float wy = ys - (float)y0, wx = xs - (float)x0;
  const float* p = de + (size_t)rc * 36 * 36;
  float a  = p[y0 * 36 + x0],       bb = p[y0 * 36 + x0 + 1];
  float cc = p[(y0 + 1) * 36 + x0], dd = p[(y0 + 1) * 36 + x0 + 1];
  float top = a  * (1.0f - wx) + bb * wx;
  float bot = cc * (1.0f - wx) + dd * wx;
  out[t] = top * (1.0f - wy) + bot * wy;
}

// 1x1 conv: (16,64,10,10) -> scale/shift [n][pix][c] channels-last
__global__ void k_dense(const float* __restrict__ resized, const float* __restrict__ dw,
                        const float* __restrict__ db, float* __restrict__ scaleA,
                        float* __restrict__ shiftA) {
  int t = blockIdx.x * 256 + threadIdx.x;  // 409600
  int pix = t % 100; int r = t / 100;      // r = n*256+cc
  int cc = r & 255; int n = r >> 8;
  float acc = db[cc];
  const float* rp = resized + (size_t)n * 64 * 100 + pix;
  const float* wp = dw + cc * 64;
  for (int cd = 0; cd < 64; ++cd) acc += wp[cd] * rp[cd * 100];
  if (cc < 128) scaleA[(n * 100 + pix) * 128 + cc] = acc;
  else          shiftA[(n * 100 + pix) * 128 + (cc - 128)] = acc;
}

// ---------------- patch extraction + GN1 + SiLU + FiLM -> Pa bf16 [p][y][x][c] ----------------

__global__ __launch_bounds__(256) void k_patch(
    const float* __restrict__ xin, const int* __restrict__ idxArr,
    const float* __restrict__ g1, const float* __restrict__ b1,
    const float* __restrict__ scaleA, const float* __restrict__ shiftA,
    unsigned short* __restrict__ Pa)
{
  __shared__ float pbuf[100 * 128];
  __shared__ float gsum[32], gsq[32];
  int p = blockIdx.x;                // 0..767
  int n = p / 48, k = p - n * 48;
  int blk = idxArr[k];
  int bi = blk / 9, bj = blk - bi * 9;
  int t = threadIdx.x;
  int c = t >> 1, hp = t & 1;
  if (t < 32) { gsum[t] = 0.f; gsq[t] = 0.f; }
  __syncthreads();
  float s = 0.f, s2 = 0.f;
  const float* xc = xin + ((size_t)(n * 128 + c)) * 72 * 72;
  for (int i = 0; i < 50; ++i) {
    int pix = hp * 50 + i;
    int y = pix / 10, x = pix - y * 10;
    int gy = bi * 8 + y - 1, gx = bj * 8 + x - 1;
    float v = 0.f;
    if (gy >= 0 && gy < 72 && gx >= 0 && gx < 72) v = xc[gy * 72 + gx];
    pbuf[pix * 128 + c] = v;
    s += v; s2 += v * v;
  }
  int g = c >> 2;
  atomicAdd(&gsum[g], s);
  atomicAdd(&gsq[g], s2);
  __syncthreads();
  float mu  = gsum[g] * (1.0f / 400.0f);
  float var = gsq[g] * (1.0f / 400.0f) - mu * mu;
  float rs  = rsqrtf(var + 1e-5f);
  float gg = g1[c], bbv = b1[c];
  for (int i = 0; i < 50; ++i) {
    int pix = hp * 50 + i;
    float v  = pbuf[pix * 128 + c];
    float yv = (v - mu) * rs * gg + bbv;
    float sl = siluf(yv);
    float fs = scaleA[(n * 100 + pix) * 128 + c];
    float fh = shiftA[(n * 100 + pix) * 128 + c];
    Pa[((size_t)p * 100 + pix) * 128 + c] = f2bf(sl * (1.0f + fs) + fh);
  }
}

// ---------------- GN2 + SiLU -> padded Pb bf16 [p][10][10][c], zero border ----------------

__global__ __launch_bounds__(256) void k_gn2(
    const float* __restrict__ H1, const float* __restrict__ g2, const float* __restrict__ b2,
    unsigned short* __restrict__ Pb)
{
  __shared__ float pbuf[64 * 128];
  __shared__ float gsum[32], gsq[32];
  int p = blockIdx.x;
  int t = threadIdx.x;
  int c = t >> 1, hp = t & 1;
  if (t < 32) { gsum[t] = 0.f; gsq[t] = 0.f; }
  for (int j = t; j < 36 * 128; j += 256) {    // zero the 36 border pixels
    int bIdx = j >> 7; int cz = j & 127;
    int y, x;
    if (bIdx < 10)      { y = 0;         x = bIdx;      }
    else if (bIdx < 20) { y = 9;         x = bIdx - 10; }
    else if (bIdx < 28) { y = bIdx - 19; x = 0;         }
    else                { y = bIdx - 27; x = 9;         }
    Pb[((size_t)p * 100 + y * 10 + x) * 128 + cz] = 0;
  }
  __syncthreads();
  float s = 0.f, s2 = 0.f;
  for (int i = 0; i < 32; ++i) {
    int pix = hp * 32 + i;
    float v = H1[((size_t)p * 64 + pix) * 128 + c];
    pbuf[pix * 128 + c] = v;
    s += v; s2 += v * v;
  }
  int g = c >> 2;
  atomicAdd(&gsum[g], s);
  atomicAdd(&gsq[g], s2);
  __syncthreads();
  float mu  = gsum[g] * (1.0f / 256.0f);
  float var = gsq[g] * (1.0f / 256.0f) - mu * mu;
  float rs  = rsqrtf(var + 1e-5f);
  float gg = g2[c], bbv = b2[c];
  for (int i = 0; i < 32; ++i) {
    int pix = hp * 32 + i;
    float v  = pbuf[pix * 128 + c];
    float yv = (v - mu) * rs * gg + bbv;
    float sl = siluf(yv);
    int oy = pix >> 3, ox = pix & 7;
    Pb[((size_t)p * 100 + (oy + 1) * 10 + (ox + 1)) * 128 + c] = f2bf(sl);
  }
}

// ---------------- implicit-GEMM conv via v_wmma_f32_16x16x32_bf16 ----------------
// M = 49152 (patch*64 pixels), N = 128 (Co), K = 1152 = 9 taps x 128 ch
// Tiles staged with GLOBAL_LOAD_ASYNC_TO_LDS_B128 (ASYNCcnt), double-buffered LDS.
// EPI 0: H1[p][pix][co] = acc + conv_in_b[co] + emb_out[n][co]
// EPI 1: out[n][co][gy][gx] = x[...] + acc + conv_out_b[co]   (scatter to idx block)

template <int EPI>
__global__ __launch_bounds__(256) void k_gemm(
    const unsigned short* __restrict__ Aact, const unsigned short* __restrict__ Wt,
    const float* __restrict__ bias, const float* __restrict__ embOut,
    float* __restrict__ H1,
    const float* __restrict__ xin, float* __restrict__ dout,
    const int* __restrict__ idxArr)
{
  __shared__ unsigned short Alds[2][128 * 40];   // 128 rows x 32 K bf16, pad to 40 (80B row)
  __shared__ unsigned short Blds[2][128 * 40];   // 128 cols(N) x 32 K bf16, pad to 40
  int t    = threadIdx.x;
  int lane = t & 31;
  int wv   = t >> 5;
  int wm   = (wv >> 1) * 32;      // wave M offset (4 waves down M)
  int wn   = (wv & 1) * 64;       // wave N offset (2 waves across N)
  int hlf  = lane >> 4;
  int l16  = lane & 15;
  int mBase = blockIdx.x * 128;
  int rA = t >> 1;                // loader: row 0..127
  int cA = (t & 1) * 16;          // loader: K col 0 or 16
  int mg = mBase + rA;
  int p_ld   = mg >> 6;
  int pix_ld = mg & 63;
  int oy = pix_ld >> 3, ox = pix_ld & 7;

  // LDS byte offsets for this thread's async-load destinations (generic addr low 32 = LDS offset)
  unsigned ldsAbase = (unsigned)(uintptr_t)(&Alds[0][0]) + (unsigned)(rA * 80 + cA * 2);
  unsigned ldsBbase = (unsigned)(uintptr_t)(&Blds[0][0]) + (unsigned)(rA * 80 + cA * 2);

  auto issueTile = [&](int kt, int buf) {
    int tap  = kt >> 2;
    int cblk = (kt & 3) * 32;
    int ky = tap / 3;
    int kx = tap - ky * 3;
    unsigned long long ga = (unsigned long long)(uintptr_t)(Aact +
        ((size_t)p_ld * 100 + (size_t)(oy + ky) * 10 + (ox + kx)) * 128 + cblk + cA);
    unsigned long long gb = (unsigned long long)(uintptr_t)(Wt +
        ((size_t)tap * 128 + rA) * 128 + cblk + cA);
    unsigned la = ldsAbase + (unsigned)(buf * 128 * 80);
    unsigned lb = ldsBbase + (unsigned)(buf * 128 * 80);
    // INST_OFFSET applies to both the global and LDS side (ISA 10.x async pseudocode),
    // so one address pair covers both 16B halves of this thread's 32B chunk.
    asm volatile("global_load_async_to_lds_b128 %0, %1, off"           :: "v"(la), "v"(ga) : "memory");
    asm volatile("global_load_async_to_lds_b128 %0, %1, off offset:16" :: "v"(la), "v"(ga) : "memory");
    asm volatile("global_load_async_to_lds_b128 %0, %1, off"           :: "v"(lb), "v"(gb) : "memory");
    asm volatile("global_load_async_to_lds_b128 %0, %1, off offset:16" :: "v"(lb), "v"(gb) : "memory");
  };

  v8f zero8 = {0.f, 0.f, 0.f, 0.f, 0.f, 0.f, 0.f, 0.f};
  v8f acc[2][4];
  for (int i = 0; i < 2; ++i)
    for (int j = 0; j < 4; ++j) acc[i][j] = zero8;

  issueTile(0, 0);                               // prologue: tile 0 -> buffer 0

  for (int kt = 0; kt < 36; ++kt) {
    int cur = kt & 1;
    if (kt + 1 < 36) {
      issueTile(kt + 1, cur ^ 1);                // overlap: next tile streams during compute
      // async loads complete in order -> cnt<=4 means tile kt's 4 loads are done
      asm volatile("s_wait_asynccnt 0x4" ::: "memory");
    } else {
      asm volatile("s_wait_asynccnt 0x0" ::: "memory");
    }
    __syncthreads();                             // tile kt visible to all waves

    const unsigned short* Ab = &Alds[cur][0];
    const unsigned short* Bb = &Blds[cur][0];
    Frag af[2], bfr[4];
    for (int fm = 0; fm < 2; ++fm) {
      int m = wm + fm * 16 + l16;
      af[fm].q[0] = *(const uint4*)&Ab[m * 40 + 8 * hlf];        // K = 8*hlf..+7
      af[fm].q[1] = *(const uint4*)&Ab[m * 40 + 16 + 8 * hlf];   // K = 16+8*hlf..+7
    }
    for (int fn = 0; fn < 4; ++fn) {
      int nn = wn + fn * 16 + l16;
      bfr[fn].q[0] = *(const uint4*)&Bb[nn * 40 + 8 * hlf];      // K = 8*hlf..+7
      bfr[fn].q[1] = *(const uint4*)&Bb[nn * 40 + 16 + 8 * hlf]; // K = 16+8*hlf..+7
    }
    for (int fm = 0; fm < 2; ++fm)
      for (int fn = 0; fn < 4; ++fn)
        acc[fm][fn] = __builtin_amdgcn_wmma_f32_16x16x32_bf16(
            false, af[fm].v, false, bfr[fn].v, (short)0, acc[fm][fn], false, false);

    __syncthreads();                             // all waves done reading buf cur
  }

  for (int fm = 0; fm < 2; ++fm) {
    int mrow0 = mBase + wm + fm * 16 + 8 * hlf;   // multiple of 8
    int p = mrow0 >> 6;
    int pixBase = mrow0 & 63;
    int nB = p / 48;
    if (EPI == 0) {
      for (int fn = 0; fn < 4; ++fn) {
        int co = wn + fn * 16 + l16;
        float badd = bias[co] + embOut[nB * 128 + co];
        for (int r = 0; r < 8; ++r)
          H1[((size_t)p * 64 + pixBase + r) * 128 + co] = acc[fm][fn][r] + badd;
      }
    } else {
      int kk = p - nB * 48;
      int blk = idxArr[kk];
      int bi = blk / 9, bj = blk - bi * 9;
      for (int fn = 0; fn < 4; ++fn) {
        int co = wn + fn * 16 + l16;
        float bb = bias[co];
        size_t chBase = ((size_t)nB * 128 + co) * 72 * 72;
        for (int r = 0; r < 8; ++r) {
          int pix = pixBase + r;
          int oyy = pix >> 3, oxx = pix & 7;
          size_t off = chBase + (size_t)(bi * 8 + oyy) * 72 + (bj * 8 + oxx);
          dout[off] = xin[off] + acc[fm][fn][r] + bb;
        }
      }
    }
  }
}

// ---------------- launcher ----------------

extern "C" void kernel_launch(void* const* d_in, const int* in_sizes, int n_in,
                              void* d_out, int out_size, void* d_ws, size_t ws_size,
                              hipStream_t stream) {
  (void)in_sizes; (void)n_in; (void)out_size; (void)ws_size;
  const float* x          = (const float*)d_in[0];
  const float* emb        = (const float*)d_in[1];
  const float* dense_e    = (const float*)d_in[2];
  const float* gn1_g      = (const float*)d_in[3];
  const float* gn1_b      = (const float*)d_in[4];
  const float* conv_in_w  = (const float*)d_in[5];
  const float* conv_in_b  = (const float*)d_in[6];
  const float* emb_w      = (const float*)d_in[7];
  const float* emb_b      = (const float*)d_in[8];
  const float* dense_w    = (const float*)d_in[9];
  const float* dense_b    = (const float*)d_in[10];
  const float* gn2_g      = (const float*)d_in[11];
  const float* gn2_b      = (const float*)d_in[12];
  const float* conv_out_w = (const float*)d_in[13];
  const float* conv_out_b = (const float*)d_in[14];
  const int*   idx        = (const int*)d_in[15];
  float* out = (float*)d_out;

  char* w = (char*)d_ws;
  unsigned short* wt_in  = (unsigned short*)(w + 0);         // 294912 B
  unsigned short* wt_out = (unsigned short*)(w + 294912);    // 294912 B
  float* embOut          = (float*)(w + 589824);             // 8192 B
  float* resized         = (float*)(w + 598016);             // 409600 B
  float* scaleA          = (float*)(w + 1007616);            // 819200 B
  float* shiftA          = (float*)(w + 1826816);            // 819200 B
  unsigned short* Pa     = (unsigned short*)(w + 2646016);   // 19660800 B (shared Pa/Pb)
  float* H1              = (float*)(w + 22306816);           // 25165824 B -> total 47472640

  k_copy<<<10368, 256, 0, stream>>>((const float4*)x, (float4*)out, 2654208);
  k_prep_w<<<1152, 256, 0, stream>>>(conv_in_w, conv_out_w, wt_in, wt_out);
  k_emb<<<8, 256, 0, stream>>>(emb, emb_w, emb_b, embOut);
  k_resize<<<400, 256, 0, stream>>>(dense_e, resized);
  k_dense<<<1600, 256, 0, stream>>>(resized, dense_w, dense_b, scaleA, shiftA);
  k_patch<<<768, 256, 0, stream>>>(x, idx, gn1_g, gn1_b, scaleA, shiftA, Pa);
  k_gemm<0><<<384, 256, 0, stream>>>(Pa, wt_in, conv_in_b, embOut, H1,
                                     nullptr, nullptr, nullptr);
  k_gn2<<<768, 256, 0, stream>>>(H1, gn2_g, gn2_b, Pa);
  k_gemm<1><<<384, 256, 0, stream>>>(Pa, wt_out, conv_out_b, nullptr, nullptr,
                                     x, out, idx);
}
